// SparseMultiHeadAttention_51685636440103
// MI455X (gfx1250) — matrix operand order repaired and checked by reading, not compiled, not run on hardware
//
#include <hip/hip_runtime.h>

typedef __attribute__((ext_vector_type(4)))  __bf16 v4bf;
typedef __attribute__((ext_vector_type(8)))  __bf16 v8bf;
typedef __attribute__((ext_vector_type(16))) __bf16 v16bf;
typedef __attribute__((ext_vector_type(8)))  float  v8f;
typedef __attribute__((ext_vector_type(4)))  float  v4f;

#define BATCH 8
#define SEQ   4096
#define HID   1024
#define NHEAD 2048              // SCALE * SEQ
#define MTOT  (BATCH * SEQ)     // 32768 GEMM rows

#define BM 128
#define BN 256
#define BK 32
#define LDSW 40                 // halfs per LDS row: 80B stride -> 16B aligned, 20-bank stride
#define A_HALFS (BM * LDSW)     // 5120 halfs
#define B_HALFS (BN * LDSW)     // 10240 halfs
#define STAGE   (A_HALFS + B_HALFS) // 15360 halfs = 30720 B per stage

// CDNA5 async global->LDS copy (ASYNCcnt-tracked, writes LDS directly).
#define ASYNC_LDS_B128(ldsoff, gaddr)                                   \
    asm volatile("global_load_async_to_lds_b128 %0, %1, off"            \
                 :: "v"(ldsoff), "v"(gaddr) : "memory")
#define WAIT_ASYNC0() asm volatile("s_wait_asynccnt 0x0" ::: "memory")

// ---------------------------------------------------------------------------
// f32 -> bf16 conversion, 4-wide
// ---------------------------------------------------------------------------
__global__ __launch_bounds__(256) void cvt_f32_bf16(const float* __restrict__ in,
                                                    __bf16* __restrict__ out,
                                                    long long n4) {
    long long i = (long long)blockIdx.x * blockDim.x + threadIdx.x;
    if (i >= n4) return;
    const v4f f = *(const v4f*)(in + i * 4);
    v4bf o;
    o[0] = (__bf16)f[0]; o[1] = (__bf16)f[1];
    o[2] = (__bf16)f[2]; o[3] = (__bf16)f[3];
    *(v4bf*)(out + i * 4) = o;
}

// ---------------------------------------------------------------------------
// Y(MxN) = A(MxK, bf16) * W(NxK, bf16)^T + bias(N, f32)
// 128x256 block tile, BK=32, 8 waves (2x4); each wave owns a 64x64 patch
// = 4x4 WMMA accumulators. Double-buffered LDS fed by async global->LDS.
// ---------------------------------------------------------------------------
template <typename OutT>
__global__ __launch_bounds__(256) void gemm_bf16_nt(const __bf16* __restrict__ A,
                                                    const __bf16* __restrict__ W,
                                                    const float*  __restrict__ bias,
                                                    OutT* __restrict__ Y,
                                                    int M, int N, int K) {
    __shared__ __bf16 smem[2 * STAGE];

    const int tid  = threadIdx.x;
    const int lane = tid & 31;
    const int wave = tid >> 5;
    const int wr   = wave >> 2;            // 0..1  (wave row: 64 rows each)
    const int wc   = wave & 3;             // 0..3  (wave col: 64 cols each)
    const int m0   = blockIdx.y * BM;
    const int n0   = blockIdx.x * BN;

    const int lm    = lane & 15;           // row within 16x16 fragment
    const int kselA = (lane >> 4) * 8;     // A: K-group 0 or 8 (ISA 7.12.2)
    const int kselB = (lane >> 4) * 8;     // B: K-group 0 or 8

    // ---- per-thread async-copy chunk coordinates (16B chunks)
    // A tile: 512 chunks (2/thread); B tile: 1024 chunks (4/thread)
    const uint32_t sbase = (uint32_t)(size_t)&smem[0];
    uint32_t lA[2]; const __bf16* gA[2];
#pragma unroll
    for (int h = 0; h < 2; ++h) {
        const int q = tid + h * 256, r = q >> 2, p = q & 3;
        lA[h] = sbase + (uint32_t)((r * LDSW + p * 8) * 2);
        gA[h] = A + (size_t)(m0 + r) * K + p * 8;
    }
    uint32_t lB[4]; const __bf16* gB[4];
#pragma unroll
    for (int h = 0; h < 4; ++h) {
        const int q = tid + h * 256, r = q >> 2, p = q & 3;
        lB[h] = sbase + (uint32_t)((A_HALFS + r * LDSW + p * 8) * 2);
        gB[h] = W + (size_t)(n0 + r) * K + p * 8;
    }

    v8f acc[4][4] = {};

    // ---- prologue: async-load tile 0 into stage 0
#pragma unroll
    for (int h = 0; h < 2; ++h) ASYNC_LDS_B128(lA[h], gA[h]);
#pragma unroll
    for (int h = 0; h < 4; ++h) ASYNC_LDS_B128(lB[h], gB[h]);

    const int ntiles = K / BK;
    for (int kt = 0; kt < ntiles; ++kt) {
        WAIT_ASYNC0();              // my wave's contribution to tile kt is in LDS
        __syncthreads();            // everyone's is

        // kick off tile kt+1 into the other stage while we compute tile kt
        if (kt + 1 < ntiles) {
            const uint32_t bo = (uint32_t)(((kt + 1) & 1) * (STAGE * 2));
            const int kh = (kt + 1) * BK;
#pragma unroll
            for (int h = 0; h < 2; ++h) ASYNC_LDS_B128(lA[h] + bo, gA[h] + kh);
#pragma unroll
            for (int h = 0; h < 4; ++h) ASYNC_LDS_B128(lB[h] + bo, gB[h] + kh);
        }

        const __bf16* sA = smem + (kt & 1) * STAGE;
        const __bf16* sB = sA + A_HALFS;

        // ---- B fragments: lane holds K=kselB..+7 and 16+kselB..+7 of column (wc*64 + nt*16 + lm)
        v16bf bfrag[4];
#pragma unroll
        for (int nt = 0; nt < 4; ++nt) {
            const __bf16* p = &sB[(wc * 64 + nt * 16 + lm) * LDSW + kselB];
            v8bf lo = *(const v8bf*)p;
            v8bf hi = *(const v8bf*)(p + 16);
            bfrag[nt] = __builtin_shufflevector(lo, hi, 0,1,2,3,4,5,6,7,
                                                        8,9,10,11,12,13,14,15);
        }
        // ---- A fragments + 16 WMMAs
#pragma unroll
        for (int mt = 0; mt < 4; ++mt) {
            const __bf16* p = &sA[(wr * 64 + mt * 16 + lm) * LDSW + kselA];
            v8bf lo = *(const v8bf*)p;           // K = kselA .. kselA+7
            v8bf hi = *(const v8bf*)(p + 16);    // K = 16+kselA .. 16+kselA+7
            v16bf afrag = __builtin_shufflevector(lo, hi, 0,1,2,3,4,5,6,7,
                                                          8,9,10,11,12,13,14,15);
#pragma unroll
            for (int nt = 0; nt < 4; ++nt) {
                acc[mt][nt] = __builtin_amdgcn_wmma_f32_16x16x32_bf16(
                    false, afrag, false, bfrag[nt], (short)0, acc[mt][nt],
                    false, false);
            }
        }
    }

    // ---- epilogue: C/D layout — VGPR r: row r (lanes 0-15) / r+8 (lanes 16-31)
    const int rsel = (lane >> 4) * 8;
#pragma unroll
    for (int nt = 0; nt < 4; ++nt) {
        const int col = n0 + wc * 64 + nt * 16 + lm;
        const float bv = bias[col];
#pragma unroll
        for (int mt = 0; mt < 4; ++mt) {
#pragma unroll
            for (int r = 0; r < 8; ++r) {
                const int row = m0 + wr * 64 + mt * 16 + rsel + r;
                Y[(size_t)row * N + col] = (OutT)(acc[mt][nt][r] + bv);
            }
        }
    }
}

// ---------------------------------------------------------------------------
// Degenerate attention: per head, 2 tokens x 1024 dim. One wave32 per head.
// Writes attn probs (B,NH,2,2) and scatters attended rows back to original
// token order (so the O-projection GEMM reads a plain row-major matrix).
// ---------------------------------------------------------------------------
__global__ __launch_bounds__(256) void attn_kernel(const __bf16* __restrict__ Q,
                                                   const __bf16* __restrict__ K,
                                                   const __bf16* __restrict__ V,
                                                   const int* __restrict__ perm,
                                                   __bf16* __restrict__ AO,
                                                   float* __restrict__ attn) {
    const int lane = threadIdx.x & 31;
    const int wave = threadIdx.x >> 5;
    const int hh   = blockIdx.x * 8 + wave;     // global (b,head) index
    const int b    = hh >> 11;                  // / NHEAD
    const int h    = hh & (NHEAD - 1);
    const int i0   = perm[h * 2];
    const int i1   = perm[h * 2 + 1];

    const size_t bb = (size_t)b * SEQ * HID;
    const __bf16* q0 = Q + bb + (size_t)i0 * HID;
    const __bf16* q1 = Q + bb + (size_t)i1 * HID;
    const __bf16* k0 = K + bb + (size_t)i0 * HID;
    const __bf16* k1 = K + bb + (size_t)i1 * HID;

    const int c0 = lane * 32;                   // 32 contiguous dims per lane
    float s00 = 0.f, s01 = 0.f, s10 = 0.f, s11 = 0.f;
#pragma unroll
    for (int u = 0; u < 32; u += 8) {
        v8bf a0 = *(const v8bf*)(q0 + c0 + u);
        v8bf a1 = *(const v8bf*)(q1 + c0 + u);
        v8bf b0 = *(const v8bf*)(k0 + c0 + u);
        v8bf b1 = *(const v8bf*)(k1 + c0 + u);
#pragma unroll
        for (int e = 0; e < 8; ++e) {
            const float fa0 = (float)a0[e], fa1 = (float)a1[e];
            const float fb0 = (float)b0[e], fb1 = (float)b1[e];
            s00 += fa0 * fb0; s01 += fa0 * fb1;
            s10 += fa1 * fb0; s11 += fa1 * fb1;
        }
    }
#pragma unroll
    for (int off = 16; off > 0; off >>= 1) {    // wave32 butterfly reduce
        s00 += __shfl_xor(s00, off, 32);
        s01 += __shfl_xor(s01, off, 32);
        s10 += __shfl_xor(s10, off, 32);
        s11 += __shfl_xor(s11, off, 32);
    }
    const float scale = 0.03125f;               // 1/sqrt(1024)
    s00 *= scale; s01 *= scale; s10 *= scale; s11 *= scale;

    const float mr0 = fmaxf(s00, s01);
    const float e00 = __expf(s00 - mr0), e01 = __expf(s01 - mr0);
    const float r0  = 1.f / (e00 + e01);
    const float p00 = e00 * r0, p01 = e01 * r0;

    const float mr1 = fmaxf(s10, s11);
    const float e10 = __expf(s10 - mr1), e11 = __expf(s11 - mr1);
    const float r1  = 1.f / (e10 + e11);
    const float p10 = e10 * r1, p11 = e11 * r1;

    if (lane == 0) {
        float* ap = attn + (size_t)hh * 4;
        ap[0] = p00; ap[1] = p01; ap[2] = p10; ap[3] = p11;
    }

    const __bf16* v0 = V + bb + (size_t)i0 * HID;
    const __bf16* v1 = V + bb + (size_t)i1 * HID;
    __bf16* o0 = AO + bb + (size_t)i0 * HID;    // scatter to original order
    __bf16* o1 = AO + bb + (size_t)i1 * HID;
#pragma unroll
    for (int u = 0; u < 32; u += 8) {
        v8bf x0 = *(const v8bf*)(v0 + c0 + u);
        v8bf x1 = *(const v8bf*)(v1 + c0 + u);
        v8bf y0, y1;
#pragma unroll
        for (int e = 0; e < 8; ++e) {
            const float f0 = (float)x0[e], f1 = (float)x1[e];
            y0[e] = (__bf16)(p00 * f0 + p01 * f1);
            y1[e] = (__bf16)(p10 * f0 + p11 * f1);
        }
        *(v8bf*)(o0 + c0 + u) = y0;
        *(v8bf*)(o1 + c0 + u) = y1;
    }
}

// ---------------------------------------------------------------------------
extern "C" void kernel_launch(void* const* d_in, const int* in_sizes, int n_in,
                              void* d_out, int out_size, void* d_ws, size_t ws_size,
                              hipStream_t stream) {
    const float* x  = (const float*)d_in[0];
    const float* Wq = (const float*)d_in[1];
    const float* bq = (const float*)d_in[2];
    const float* Wk = (const float*)d_in[3];
    const float* bk = (const float*)d_in[4];
    const float* Wv = (const float*)d_in[5];
    const float* bv = (const float*)d_in[6];
    const float* Wo = (const float*)d_in[7];
    const float* bo = (const float*)d_in[8];
    const int* perm = (const int*)d_in[9];
    float* out = (float*)d_out;

    const size_t HH = (size_t)HID * HID;        // 1,048,576
    const size_t NN = (size_t)MTOT * HID;       // 33,554,432

    __bf16* Wqb = (__bf16*)d_ws;
    __bf16* Wkb = Wqb + HH;
    __bf16* Wvb = Wkb + HH;
    __bf16* Wob = Wvb + HH;
    __bf16* xbf = Wob + HH;
    __bf16* Qb  = xbf + NN;
    __bf16* Kb  = Qb + NN;
    __bf16* Vb  = Kb + NN;
    __bf16* Ab  = xbf;                          // reuse x region for attended

    // 1) demote inputs to bf16
    cvt_f32_bf16<<<(int)(NN / 4 / 256), 256, 0, stream>>>(x,  xbf, NN / 4);
    cvt_f32_bf16<<<(int)(HH / 4 / 256), 256, 0, stream>>>(Wq, Wqb, HH / 4);
    cvt_f32_bf16<<<(int)(HH / 4 / 256), 256, 0, stream>>>(Wk, Wkb, HH / 4);
    cvt_f32_bf16<<<(int)(HH / 4 / 256), 256, 0, stream>>>(Wv, Wvb, HH / 4);
    cvt_f32_bf16<<<(int)(HH / 4 / 256), 256, 0, stream>>>(Wo, Wob, HH / 4);

    // 2) Q/K/V projections (WMMA bf16, f32 accumulate, bf16 out)
    dim3 grid(HID / BN, MTOT / BM);             // (4, 256)
    gemm_bf16_nt<__bf16><<<grid, 256, 0, stream>>>(xbf, Wqb, bq, Qb, MTOT, HID, HID);
    gemm_bf16_nt<__bf16><<<grid, 256, 0, stream>>>(xbf, Wkb, bk, Kb, MTOT, HID, HID);
    gemm_bf16_nt<__bf16><<<grid, 256, 0, stream>>>(xbf, Wvb, bv, Vb, MTOT, HID, HID);

    // 3) degenerate 2-token-per-head attention; writes probs + attended rows
    attn_kernel<<<(BATCH * NHEAD) / 8, 256, 0, stream>>>(Qb, Kb, Vb, perm, Ab,
                                                         out + NN);

    // 4) output projection straight into d_out (f32)
    gemm_bf16_nt<float><<<grid, 256, 0, stream>>>(Ab, Wob, bo, out, MTOT, HID, HID);
}
